// SpconvBasicBlock_29738353558082
// MI455X (gfx1250) — matrix operand order repaired and checked by reading, not compiled, not run on hardware
//
#include <hip/hip_runtime.h>
#include <hip/hip_bf16.h>

typedef __attribute__((ext_vector_type(16))) _Float16 v16h;
typedef __attribute__((ext_vector_type(8)))  float    v8f;

#define KCLUST 2048
#define NEGS   0.01f

__device__ __forceinline__ float lrelu_f(float v) { return v > 0.0f ? v : NEGS * v; }

__device__ __forceinline__ v8f wmma16(v16h a, v16h b, v8f c) {
  // D = A(16x32 f16) * B(32x16 f16) + C(16x16 f32)
  return __builtin_amdgcn_wmma_f32_16x16x32_f16(false, a, false, b, (short)0, c, false, false);
}

// Build the 16x32 f16 A-fragment piece this lane owns from a row-major f32 row.
// Lane 0-15 (hi=0): K = k0..k0+7 and k0+16..k0+23 ; lanes 16-31 (hi=1): +8.
__device__ __forceinline__ v16h afrag(const float* __restrict__ row, int k0) {
  float4 f0 = *(const float4*)(row + k0);
  float4 f1 = *(const float4*)(row + k0 + 4);
  float4 f2 = *(const float4*)(row + k0 + 16);
  float4 f3 = *(const float4*)(row + k0 + 20);
  v16h a;
  a[0]=(_Float16)f0.x;  a[1]=(_Float16)f0.y;  a[2]=(_Float16)f0.z;  a[3]=(_Float16)f0.w;
  a[4]=(_Float16)f1.x;  a[5]=(_Float16)f1.y;  a[6]=(_Float16)f1.z;  a[7]=(_Float16)f1.w;
  a[8]=(_Float16)f2.x;  a[9]=(_Float16)f2.y;  a[10]=(_Float16)f2.z; a[11]=(_Float16)f2.w;
  a[12]=(_Float16)f3.x; a[13]=(_Float16)f3.y; a[14]=(_Float16)f3.z; a[15]=(_Float16)f3.w;
  return a;
}

// ---- weight pre-pack: f32 [nmats][64][64] row-major -> f16 B-fragment layout ----
// packed index = m*4096 + ((ct*2+kt)*32 + lane)*16 + e
//   col = ct*16 + (lane&15), k = kt*32 + (lane>>4)*8 + (e&7) + (e>>3)*16
__global__ void pack_w64(const float* __restrict__ W, _Float16* __restrict__ out, int nmats) {
  int total = nmats << 12;
  for (int id = blockIdx.x * blockDim.x + threadIdx.x; id < total; id += gridDim.x * blockDim.x) {
    int m    = id >> 12;
    int r    = id & 4095;
    int e    = r & 15;
    int lane = (r >> 4) & 31;
    int kt   = (r >> 9) & 1;
    int ct   = (r >> 10) & 3;
    int col  = ct * 16 + (lane & 15);
    int k    = kt * 32 + (lane >> 4) * 8 + (e & 7) + (e >> 3) * 16;
    out[id] = (_Float16)W[(m * 64 + k) * 64 + col];
  }
}

// ---- GEMM: Out[N,64] = A[N,64] @ Wp(packed 64x64) (+ Cin) ----
__global__ void __launch_bounds__(256, 1)
gemm64_wmma(const float* __restrict__ A, const _Float16* __restrict__ Wp,
            const float* __restrict__ Cin, float* __restrict__ Out, int n) {
  int lane = threadIdx.x & 31;
  int tile = blockIdx.x * (blockDim.x >> 5) + (threadIdx.x >> 5);
  int row0 = tile * 16;
  if (row0 >= n) return;
  int l = lane & 15, hi = lane >> 4;
  int rr = row0 + l; if (rr >= n) rr = n - 1;
  const float* row = A + (size_t)rr * 64;
  const bool full = (row0 + 16 <= n);

  // hoist all B fragments (16 x b128 -> one clause, single wait)
  v16h b[4][2];
#pragma unroll
  for (int ct = 0; ct < 4; ++ct)
#pragma unroll
    for (int kt = 0; kt < 2; ++kt)
      b[ct][kt] = *(const v16h*)(Wp + ((size_t)(ct * 2 + kt) * 32 + lane) * 16);

  v16h a0 = afrag(row, hi * 8);
  v16h a1 = afrag(row, 32 + hi * 8);

  v8f acc[4] = {{}, {}, {}, {}};
  if (Cin) {
    if (full) {
#pragma unroll
      for (int ct = 0; ct < 4; ++ct)
#pragma unroll
        for (int j = 0; j < 8; ++j)
          acc[ct][j] = Cin[(size_t)(row0 + hi * 8 + j) * 64 + ct * 16 + l];
    } else {
#pragma unroll
      for (int ct = 0; ct < 4; ++ct)
#pragma unroll
        for (int j = 0; j < 8; ++j) {
          int r = row0 + hi * 8 + j;
          acc[ct][j] = (r < n) ? Cin[(size_t)r * 64 + ct * 16 + l] : 0.0f;
        }
    }
  }

#pragma unroll
  for (int ct = 0; ct < 4; ++ct) {
    acc[ct] = wmma16(a0, b[ct][0], acc[ct]);
    acc[ct] = wmma16(a1, b[ct][1], acc[ct]);
  }

  if (full) {
#pragma unroll
    for (int ct = 0; ct < 4; ++ct)
#pragma unroll
      for (int j = 0; j < 8; ++j)
        Out[(size_t)(row0 + hi * 8 + j) * 64 + ct * 16 + l] = acc[ct][j];
  } else {
#pragma unroll
    for (int ct = 0; ct < 4; ++ct)
#pragma unroll
      for (int j = 0; j < 8; ++j) {
        int r = row0 + hi * 8 + j;
        if (r < n) Out[(size_t)r * 64 + ct * 16 + l] = acc[ct][j];
      }
  }
}

// ---- submanifold 3^3 conv: Out[n] = sum_k gather(F, nbr[k][n]) @ Wp[k] ----
// Weights double-buffered through LDS (staged once per block), A-gather software-pipelined.
__global__ void __launch_bounds__(256, 1)
conv27_wmma(const float* __restrict__ F, const _Float16* __restrict__ Wp,
            const int* __restrict__ nbr, float* __restrict__ Out, int n) {
  __shared__ _Float16 lb[2][4096];   // 2 x 8KB weight tiles
  const int tid  = threadIdx.x;
  const int lane = tid & 31;
  const int tile = blockIdx.x * (blockDim.x >> 5) + (tid >> 5);
  const int row0 = tile * 16;
  const bool active = row0 < n;
  const int l = lane & 15, hi = lane >> 4;
  int rsel = row0 + l; if (rsel >= n) rsel = n - 1;
  const bool rowInRange = active && (row0 + l) < n;

  // stage k=0 weights (each thread moves 32B)
  ((v16h*)lb[0])[tid] = ((const v16h*)Wp)[tid];
  __syncthreads();

  // prime A for k=0
  v16h a0 = {}, a1 = {};
  {
    int idx = rowInRange ? nbr[rsel] : -1;
    if (idx >= 0) {
      const float* row = F + (size_t)idx * 64;
      a0 = afrag(row, hi * 8);
      a1 = afrag(row, 32 + hi * 8);
    }
  }

  v8f acc[4] = {{}, {}, {}, {}};
  for (int k = 0; k < 27; ++k) {
    const int cur = k & 1, nxt = cur ^ 1;
    // stage next weight tile into the other LDS buffer (overlaps compute)
    if (k < 26) {
      ((v16h*)lb[nxt])[tid] = ((const v16h*)(Wp + (size_t)(k + 1) * 4096))[tid];
      if (k < 25) __builtin_prefetch(Wp + (size_t)(k + 2) * 4096, 0, 1);
    }
    // pipeline the random gather for k+1 past this iteration's WMMAs
    v16h na0 = {}, na1 = {};
    if (k < 26) {
      int nidx = rowInRange ? nbr[(size_t)(k + 1) * n + rsel] : -1;
      if (nidx >= 0) {
        const float* row = F + (size_t)nidx * 64;
        na0 = afrag(row, hi * 8);
        na1 = afrag(row, 32 + hi * 8);
      }
    }
    const _Float16* wb = lb[cur];
#pragma unroll
    for (int ct = 0; ct < 4; ++ct) {
      v16h b0 = *(const v16h*)(wb + ((size_t)(ct * 2 + 0) * 32 + lane) * 16);
      v16h b1 = *(const v16h*)(wb + ((size_t)(ct * 2 + 1) * 32 + lane) * 16);
      acc[ct] = wmma16(a0, b0, acc[ct]);
      acc[ct] = wmma16(a1, b1, acc[ct]);
    }
    a0 = na0; a1 = na1;
    __syncthreads();   // staged nxt ready; everyone done reading cur
  }

  if (active) {
    if (row0 + 16 <= n) {
#pragma unroll
      for (int ct = 0; ct < 4; ++ct)
#pragma unroll
        for (int j = 0; j < 8; ++j)
          Out[(size_t)(row0 + hi * 8 + j) * 64 + ct * 16 + l] = acc[ct][j];
    } else {
#pragma unroll
      for (int ct = 0; ct < 4; ++ct)
#pragma unroll
        for (int j = 0; j < 8; ++j) {
          int r = row0 + hi * 8 + j;
          if (r < n) Out[(size_t)r * 64 + ct * 16 + l] = acc[ct][j];
        }
    }
  }
}

// ---- small glue kernels ----
__global__ void fill_f32(float* __restrict__ p, float v, int cnt) {
  for (int i = blockIdx.x * blockDim.x + threadIdx.x; i < cnt; i += gridDim.x * blockDim.x) p[i] = v;
}

__global__ void colstats64(const float* __restrict__ X, float* __restrict__ sum,
                           float* __restrict__ sumsq, int n) {
  int ch = threadIdx.x;   // blockDim = 64
  float s = 0.0f, q = 0.0f;
  for (int r = blockIdx.x; r < n; r += gridDim.x) {
    float v = X[(size_t)r * 64 + ch];
    s += v; q += v * v;
  }
  atomicAdd(&sum[ch], s);
  atomicAdd(&sumsq[ch], q);
}

__global__ void bn_finalize(const float* __restrict__ sum, const float* __restrict__ sumsq,
                            const float* __restrict__ g, const float* __restrict__ b,
                            float* __restrict__ scale, float* __restrict__ shift, int n) {
  int ch = threadIdx.x;   // one block of 64
  float inv = 1.0f / (float)n;
  float m = sum[ch] * inv;
  float v = sumsq[ch] * inv - m * m;
  v = fmaxf(v, 0.0f);
  float sc = g[ch] * rsqrtf(v + 1e-5f);
  scale[ch] = sc;
  shift[ch] = b[ch] - m * sc;
}

// mode 0: lrelu(bn(x)); mode 1: lrelu(bn(x)) + res; mode 2: lrelu(bn(x) + res)
__global__ void bn_apply(const float* __restrict__ X, const float* __restrict__ scale,
                         const float* __restrict__ shift, const float* __restrict__ res,
                         float* __restrict__ Out, int cnt, int mode) {
  for (int i = blockIdx.x * blockDim.x + threadIdx.x; i < cnt; i += gridDim.x * blockDim.x) {
    int ch = i & 63;
    float v = fmaf(X[i], scale[ch], shift[ch]);
    if (mode == 0)      v = lrelu_f(v);
    else if (mode == 1) v = lrelu_f(v) + res[i];
    else                v = lrelu_f(v + res[i]);
    Out[i] = v;
  }
}

__global__ void seg_count(const int* __restrict__ c, float* __restrict__ cnt, int n) {
  for (int i = blockIdx.x * blockDim.x + threadIdx.x; i < n; i += gridDim.x * blockDim.x)
    atomicAdd(&cnt[c[i]], 1.0f);
}

__global__ void seg_sum64(const float* __restrict__ X, const int* __restrict__ c,
                          float* __restrict__ seg, int cnt) {
  for (int i = blockIdx.x * blockDim.x + threadIdx.x; i < cnt; i += gridDim.x * blockDim.x)
    atomicAdd(&seg[(size_t)c[i >> 6] * 64 + (i & 63)], X[i]);
}

__global__ void seg_center(float* __restrict__ X, const int* __restrict__ c,
                           const float* __restrict__ seg, const float* __restrict__ cnt, int total) {
  for (int i = blockIdx.x * blockDim.x + threadIdx.x; i < total; i += gridDim.x * blockDim.x) {
    int cc = c[i >> 6];
    X[i] -= seg[(size_t)cc * 64 + (i & 63)] / fmaxf(cnt[cc], 1.0f);
  }
}

__global__ void seg_div(float* __restrict__ X, const int* __restrict__ c,
                        const float* __restrict__ seg, int total) {
  for (int i = blockIdx.x * blockDim.x + threadIdx.x; i < total; i += gridDim.x * blockDim.x)
    X[i] /= (seg[(size_t)c[i >> 6] * 64 + (i & 63)] + 1e-6f);
}

__device__ __forceinline__ void atomicMaxF(float* addr, float v) {
  if (!signbit(v)) atomicMax((int*)addr, __float_as_int(v));
  else             atomicMin((unsigned int*)addr, __float_as_uint(v));
}

__global__ void reduce_max(const float* __restrict__ X, float* __restrict__ gmax, int cnt) {
  float m = -3.4e38f;
  for (int i = blockIdx.x * blockDim.x + threadIdx.x; i < cnt; i += gridDim.x * blockDim.x)
    m = fmaxf(m, X[i]);
  atomicMaxF(gmax, m);
}

__global__ void exp_sub(float* __restrict__ X, const float* __restrict__ gmax, int cnt) {
  float m = gmax[0];
  for (int i = blockIdx.x * blockDim.x + threadIdx.x; i < cnt; i += gridDim.x * blockDim.x)
    X[i] = expf(X[i] - m);
}

__global__ void mul_inplace(float* __restrict__ X, const float* __restrict__ Y, int cnt) {
  for (int i = blockIdx.x * blockDim.x + threadIdx.x; i < cnt; i += gridDim.x * blockDim.x)
    X[i] *= Y[i];
}

__global__ void adp_softmax(const float* __restrict__ feat, const float* __restrict__ aW,
                            float* __restrict__ adp, int n) {
  for (int i = blockIdx.x * blockDim.x + threadIdx.x; i < n; i += gridDim.x * blockDim.x) {
    const float* row = feat + (size_t)i * 64;
    float a0 = 0.0f, a1 = 0.0f, a2 = 0.0f;
#pragma unroll
    for (int ch = 0; ch < 64; ++ch) {
      float v = row[ch];
      a0 = fmaf(v, aW[ch * 3 + 0], a0);
      a1 = fmaf(v, aW[ch * 3 + 1], a1);
      a2 = fmaf(v, aW[ch * 3 + 2], a2);
    }
    float m = fmaxf(a0, fmaxf(a1, a2));
    float e0 = expf(a0 - m), e1 = expf(a1 - m), e2 = expf(a2 - m);
    float inv = 1.0f / (e0 + e1 + e2);
    adp[(size_t)i * 3 + 0] = e0 * inv;
    adp[(size_t)i * 3 + 1] = e1 * inv;
    adp[(size_t)i * 3 + 2] = e2 * inv;
  }
}

__global__ void feats_accum(float* __restrict__ acc, const float* __restrict__ adp, int level,
                            const int* __restrict__ c, const float* __restrict__ seg, int total) {
  for (int i = blockIdx.x * blockDim.x + threadIdx.x; i < total; i += gridDim.x * blockDim.x) {
    int nidx = i >> 6;
    acc[i] += adp[(size_t)nidx * 3 + level] * seg[(size_t)c[nidx] * 64 + (i & 63)];
  }
}

// -------------------------------------------------------------------------
extern "C" void kernel_launch(void* const* d_in, const int* in_sizes, int n_in,
                              void* d_out, int out_size, void* d_ws, size_t ws_size,
                              hipStream_t stream) {
  const float* feat   = (const float*)d_in[0];
  const float* lw_W   = (const float*)d_in[1];
  const float* lw_g   = (const float*)d_in[2];
  const float* lw_b   = (const float*)d_in[3];
  const float* w_W    = (const float*)d_in[4];
  const float* proj_W = (const float*)d_in[5];
  const float* proj_g = (const float*)d_in[6];
  const float* proj_b = (const float*)d_in[7];
  const float* adW    = (const float*)d_in[8];
  const float* fuse_W = (const float*)d_in[9];
  const float* fuse_g = (const float*)d_in[10];
  const float* fuse_b = (const float*)d_in[11];
  const float* c1_W   = (const float*)d_in[12];
  const float* bn1_g  = (const float*)d_in[13];
  const float* bn1_b  = (const float*)d_in[14];
  const float* c2_W   = (const float*)d_in[15];
  const float* bn2_g  = (const float*)d_in[16];
  const float* bn2_b  = (const float*)d_in[17];
  const int*   clus   = (const int*)d_in[18];
  const int*   nbr    = (const int*)d_in[19];
  float* out = (float*)d_out;

  const int N  = in_sizes[0] / 64;
  const int NE = N * 64;

  // workspace carve-up
  char* wsb = (char*)d_ws;
  size_t off = 0;
  auto carve = [&](size_t bytes) -> char* {
    char* p = wsb + off;
    off += (bytes + 255) & ~(size_t)255;
    return p;
  };
  _Float16* Wpack = (_Float16*)carve((size_t)66 * 4096 * sizeof(_Float16));
  float* A1 = (float*)carve((size_t)NE * 4);
  float* A2 = (float*)carve((size_t)NE * 4);
  float* A3 = (float*)carve((size_t)NE * 4);   // feats accumulator
  float* A4 = (float*)carve((size_t)NE * 4);   // h (fused residual)
  float* seg     = (float*)carve((size_t)KCLUST * 64 * 4);
  float* cnt     = (float*)carve((size_t)KCLUST * 4);
  float* colsum  = (float*)carve(64 * 4);
  float* colsq   = (float*)carve(64 * 4);
  float* scale   = (float*)carve(64 * 4);
  float* shift   = (float*)carve(64 * 4);
  float* gmax    = (float*)carve(64 * 4);
  float* adp     = (float*)carve((size_t)N * 3 * 4);

  _Float16* Wp_lw = Wpack + (size_t)0  * 4096;
  _Float16* Wp_w  = Wpack + (size_t)3  * 4096;
  _Float16* Wp_pj = Wpack + (size_t)6  * 4096;
  _Float16* Wp_fu = Wpack + (size_t)10 * 4096;
  _Float16* Wp_c1 = Wpack + (size_t)12 * 4096;
  _Float16* Wp_c2 = Wpack + (size_t)39 * 4096;

  const int T = 256;
  const int gElem  = (NE + T - 1) / T;
  const int gRow   = (N + T - 1) / T;
  const int gTiles = (((N + 15) / 16) + 7) / 8;   // 8 waves (16-row tiles) per 256-thread block

  auto stats = [&](const float* X, const float* g, const float* b) {
    fill_f32<<<1, 64, 0, stream>>>(colsum, 0.0f, 64);
    fill_f32<<<1, 64, 0, stream>>>(colsq, 0.0f, 64);
    colstats64<<<256, 64, 0, stream>>>(X, colsum, colsq, N);
    bn_finalize<<<1, 64, 0, stream>>>(colsum, colsq, g, b, scale, shift, N);
  };

  // ---- weight packing (f32 -> f16 B-fragment layout) ----
  pack_w64<<<48, 256, 0, stream>>>(lw_W,   Wp_lw, 3);
  pack_w64<<<48, 256, 0, stream>>>(w_W,    Wp_w,  3);
  pack_w64<<<64, 256, 0, stream>>>(proj_W, Wp_pj, 4);
  pack_w64<<<32, 256, 0, stream>>>(fuse_W, Wp_fu, 2);
  pack_w64<<<432, 256, 0, stream>>>(c1_W,  Wp_c1, 27);
  pack_w64<<<432, 256, 0, stream>>>(c2_W,  Wp_c2, 27);

  // ---- adaptive weights + feats accumulator init ----
  adp_softmax<<<gRow, T, 0, stream>>>(feat, adW, adp, N);
  fill_f32<<<gElem, T, 0, stream>>>(A3, 0.0f, NE);

  // ---- three cluster levels ----
  for (int i = 0; i < 3; ++i) {
    const int* c = clus + (size_t)i * N;
    // pw = lrelu(bn(feat @ lw_W[i]))
    gemm64_wmma<<<gTiles, T, 0, stream>>>(feat, Wp_lw + (size_t)i * 4096, nullptr, A1, N);
    stats(A1, lw_g + i * 64, lw_b + i * 64);
    bn_apply<<<gElem, T, 0, stream>>>(A1, scale, shift, nullptr, A1, NE, 0);
    // center by cluster mean
    fill_f32<<<(KCLUST + T - 1) / T, T, 0, stream>>>(cnt, 0.0f, KCLUST);
    fill_f32<<<(KCLUST * 64 + T - 1) / T, T, 0, stream>>>(seg, 0.0f, KCLUST * 64);
    seg_count<<<gRow, T, 0, stream>>>(c, cnt, N);
    seg_sum64<<<gElem, T, 0, stream>>>(A1, c, seg, NE);
    seg_center<<<gElem, T, 0, stream>>>(A1, c, seg, cnt, NE);
    // pw @ w_W[i]; stabilized exp; per-cluster softmax
    gemm64_wmma<<<gTiles, T, 0, stream>>>(A1, Wp_w + (size_t)i * 4096, nullptr, A2, N);
    fill_f32<<<1, 64, 0, stream>>>(gmax, -3.4e38f, 1);
    reduce_max<<<gElem, T, 0, stream>>>(A2, gmax, NE);
    exp_sub<<<gElem, T, 0, stream>>>(A2, gmax, NE);
    fill_f32<<<(KCLUST * 64 + T - 1) / T, T, 0, stream>>>(seg, 0.0f, KCLUST * 64);
    seg_sum64<<<gElem, T, 0, stream>>>(A2, c, seg, NE);
    seg_div<<<gElem, T, 0, stream>>>(A2, c, seg, NE);
    // pfeat = lrelu(bn(feat @ proj_W[i])) * pw ; pool ; weighted broadcast into A3
    gemm64_wmma<<<gTiles, T, 0, stream>>>(feat, Wp_pj + (size_t)i * 4096, nullptr, A1, N);
    stats(A1, proj_g + i * 64, proj_b + i * 64);
    bn_apply<<<gElem, T, 0, stream>>>(A1, scale, shift, nullptr, A1, NE, 0);
    mul_inplace<<<gElem, T, 0, stream>>>(A1, A2, NE);
    fill_f32<<<(KCLUST * 64 + T - 1) / T, T, 0, stream>>>(seg, 0.0f, KCLUST * 64);
    seg_sum64<<<gElem, T, 0, stream>>>(A1, c, seg, NE);
    feats_accum<<<gElem, T, 0, stream>>>(A3, adp, i, c, seg, NE);
  }

  // ---- f = lrelu(bn(feat @ proj_W[3])) ----
  gemm64_wmma<<<gTiles, T, 0, stream>>>(feat, Wp_pj + (size_t)3 * 4096, nullptr, A1, N);
  stats(A1, proj_g + 3 * 64, proj_b + 3 * 64);
  bn_apply<<<gElem, T, 0, stream>>>(A1, scale, shift, nullptr, A1, NE, 0);

  // ---- fuse: [f | feats] @ fuse_W == f @ W[0:64] + feats @ W[64:128] ----
  gemm64_wmma<<<gTiles, T, 0, stream>>>(A1, Wp_fu, nullptr, A4, N);
  gemm64_wmma<<<gTiles, T, 0, stream>>>(A3, Wp_fu + 4096, A4, A4, N);
  stats(A4, fuse_g, fuse_b);
  bn_apply<<<gElem, T, 0, stream>>>(A4, scale, shift, feat, A4, NE, 1);  // h = lrelu(bn(.)) + feat

  // ---- conv1 -> bn1 -> lrelu  (v1 lives in d_out) ----
  conv27_wmma<<<gTiles, T, 0, stream>>>(A4, Wp_c1, nbr, A1, N);
  stats(A1, bn1_g, bn1_b);
  bn_apply<<<gElem, T, 0, stream>>>(A1, scale, shift, nullptr, out, NE, 0);

  // ---- conv2 -> bn2 ; out = lrelu(v + h) ----
  conv27_wmma<<<gTiles, T, 0, stream>>>(out, Wp_c2, nbr, A2, N);
  stats(A2, bn2_g, bn2_b);
  bn_apply<<<gElem, T, 0, stream>>>(A2, scale, shift, A4, out, NE, 2);
}
